// LSTMModel_49804440765052
// MI455X (gfx1250) — matrix operand order repaired and checked by reading, not compiled, not run on hardware
//
#include <hip/hip_runtime.h>

#define B_SZ 256
#define T_SZ 256
#define D_SZ 64
#define H_SZ 512
#define F_SZ 1024
#define N_SZ 64

typedef __bf16 bf16_t;
typedef __attribute__((ext_vector_type(16))) __bf16 v16bf;
typedef __attribute__((ext_vector_type(8)))  __bf16 v8bf;
typedef __attribute__((ext_vector_type(8)))  float  v8f;

__device__ __forceinline__ unsigned short f2bf(float f) {
  unsigned int u = __float_as_uint(f);
  unsigned int r = u + 0x7fffu + ((u >> 16) & 1u);   // round-to-nearest-even
  return (unsigned short)(r >> 16);
}

// ---- WMMA fragment loaders (per CDNA5 ISA 7.12.2 layouts, wave32) ----
// A (16x32 bf16, row-major, stride lda): lane m=lane&15; lanes<16 hold K 0..7 &
// 16..23, lanes>=16 hold K 8..15 & 24..31.
__device__ __forceinline__ v16bf ld_frag_a(const bf16_t* base, int lda, int lane) {
  int m  = lane & 15;
  int kb = (lane >> 4) << 3;           // 0 or 8
  const bf16_t* p = base + (size_t)m * lda + kb;
  v16bf f;
  ((v8bf*)&f)[0] = *(const v8bf*)(p);
  ((v8bf*)&f)[1] = *(const v8bf*)(p + 16);
  return f;
}
// B (32x16 bf16). We store B as rows of B^T: bt[n][k] = B[k][n]. Lane holds
// column n=lane&15; lanes<16: K 0..7 & 16..23, lanes>=16: K 8..15 & 24..31.
__device__ __forceinline__ v16bf ld_frag_b(const bf16_t* bt, int ldb, int lane) {
  int n  = lane & 15;
  int kb = (lane >> 4) << 3;           // 0 or 8
  const bf16_t* p = bt + (size_t)n * ldb + kb;
  v16bf f;
  ((v8bf*)&f)[0] = *(const v8bf*)(p);
  ((v8bf*)&f)[1] = *(const v8bf*)(p + 16);
  return f;
}
__device__ __forceinline__ v8f wmma_bf16(v16bf a, v16bf b, v8f c) {
  return __builtin_amdgcn_wmma_f32_16x16x32_bf16(false, a, false, b, (short)0, c,
                                                 false, false);
}

// ---------------- conversion / init kernels ----------------
__global__ void k_f32_to_bf16(const float* __restrict__ in,
                              unsigned short* __restrict__ out, int n) {
  int i = blockIdx.x * blockDim.x + threadIdx.x;
  if (i < n) out[i] = f2bf(in[i]);
}

// qst[b][j][i] = Q_sqrt[b][i][j]  (bf16)
__global__ void k_qs_transpose_bf16(const float* __restrict__ qs,
                                    unsigned short* __restrict__ out, int n) {
  int i = blockIdx.x * blockDim.x + threadIdx.x;
  if (i >= n) return;
  int b = i >> 12;
  int j = (i >> 6) & 63;
  int col = i & 63;
  out[i] = f2bf(qs[((size_t)b << 12) + ((size_t)col << 6) + j]);
}

__global__ void k_init_state(unsigned short* __restrict__ h0,
                             float* __restrict__ c, int n) {
  int i = blockIdx.x * blockDim.x + threadIdx.x;
  if (i < n) { h0[i] = 0; c[i] = 0.f; }
}

// ---------------- LSTM step: gates = x_t@Wih^T + h@Whh^T + bias ----------------
// grid: 64 blocks x 256 thr = 512 waves; wave -> (mt in [0,16), nt in [0,32))
__global__ void k_lstm_step(const bf16_t* __restrict__ xbf,
                            const bf16_t* __restrict__ wih,
                            const bf16_t* __restrict__ whh,
                            const float* __restrict__ b_ih,
                            const float* __restrict__ b_hh,
                            const bf16_t* __restrict__ hin,
                            bf16_t* __restrict__ hout,
                            float* __restrict__ c, int t) {
  int lane = threadIdx.x & 31;
  int wave = blockIdx.x * (blockDim.x >> 5) + (threadIdx.x >> 5);
  int mt = wave >> 5;
  int nt = wave & 31;
  int row0 = mt << 4;
  int hcol0 = nt << 4;

  v8f acc[4];
  for (int g = 0; g < 4; ++g)
    for (int r = 0; r < 8; ++r) acc[g][r] = 0.f;

  for (int g = 0; g < 4; ++g) {
    int gc0 = (g << 9) + hcol0;
    // x_t @ W_ih^T  (K = 64)
    const bf16_t* Ax = xbf + ((size_t)row0 * T_SZ + t) * D_SZ;
    const bf16_t* Bx = wih + (size_t)gc0 * D_SZ;
    for (int kk = 0; kk < D_SZ; kk += 32) {
      v16bf a = ld_frag_a(Ax + kk, T_SZ * D_SZ, lane);
      v16bf b = ld_frag_b(Bx + kk, D_SZ, lane);
      acc[g] = wmma_bf16(a, b, acc[g]);
    }
    // h @ W_hh^T  (K = 512)
    const bf16_t* Ah = hin + (size_t)row0 * H_SZ;
    const bf16_t* Bh = whh + (size_t)gc0 * H_SZ;
    for (int kk = 0; kk < H_SZ; kk += 32) {
      v16bf a = ld_frag_a(Ah + kk, H_SZ, lane);
      v16bf b = ld_frag_b(Bh + kk, H_SZ, lane);
      acc[g] = wmma_bf16(a, b, acc[g]);
    }
  }

  int n = lane & 15;
  int colh = hcol0 + n;
  int rb = (lane >> 4) << 3;
  float bs[4];
  for (int g = 0; g < 4; ++g) bs[g] = b_ih[(g << 9) + colh] + b_hh[(g << 9) + colh];
  for (int r = 0; r < 8; ++r) {
    int row = row0 + rb + r;
    size_t ci = (size_t)row * H_SZ + colh;
    float iv = 1.f / (1.f + __expf(-(acc[0][r] + bs[0])));
    float fv = 1.f / (1.f + __expf(-(acc[1][r] + bs[1])));
    float gv = tanhf(acc[2][r] + bs[2]);
    float ov = 1.f / (1.f + __expf(-(acc[3][r] + bs[3])));
    float cv = fv * c[ci] + iv * gv;
    c[ci] = cv;
    ((unsigned short*)hout)[ci] = f2bf(ov * tanhf(cv));
  }
}

// ---------------- fc1: [256,512] x [512,1024] + leaky ----------------
__global__ void k_fc1(const bf16_t* __restrict__ h, const bf16_t* __restrict__ w,
                      const float* __restrict__ bias,
                      unsigned short* __restrict__ out) {
  int lane = threadIdx.x & 31;
  int wave = blockIdx.x * 8 + (threadIdx.x >> 5);
  int mt = wave >> 6;          // 16 row tiles
  int nt = wave & 63;          // 64 col tiles
  int row0 = mt << 4, col0 = nt << 4;
  v8f acc;
  for (int r = 0; r < 8; ++r) acc[r] = 0.f;
  const bf16_t* A = h + (size_t)row0 * H_SZ;
  const bf16_t* Bw = w + (size_t)col0 * H_SZ;
  for (int kk = 0; kk < H_SZ; kk += 32)
    acc = wmma_bf16(ld_frag_a(A + kk, H_SZ, lane), ld_frag_b(Bw + kk, H_SZ, lane), acc);
  int n = lane & 15, rb = (lane >> 4) << 3;
  float bv = bias[col0 + n];
  for (int r = 0; r < 8; ++r) {
    float v = acc[r] + bv;
    v = (v > 0.f) ? v : (-0.1f) * v;   // NEG_SLOPE = -0.1
    out[(size_t)(row0 + rb + r) * F_SZ + col0 + n] = f2bf(v);
  }
}

// ---------------- fc2: [256,1024] x [1024,64] -> f32 logits ----------------
__global__ void k_fc2(const bf16_t* __restrict__ a, const bf16_t* __restrict__ w,
                      const float* __restrict__ bias, float* __restrict__ out) {
  int lane = threadIdx.x & 31;
  int wave = blockIdx.x * 8 + (threadIdx.x >> 5);
  int mt = wave >> 2;          // 16 row tiles
  int nt = wave & 3;           // 4 col tiles
  int row0 = mt << 4, col0 = nt << 4;
  v8f acc;
  for (int r = 0; r < 8; ++r) acc[r] = 0.f;
  const bf16_t* A = a + (size_t)row0 * F_SZ;
  const bf16_t* Bw = w + (size_t)col0 * F_SZ;
  for (int kk = 0; kk < F_SZ; kk += 32)
    acc = wmma_bf16(ld_frag_a(A + kk, F_SZ, lane), ld_frag_b(Bw + kk, F_SZ, lane), acc);
  int n = lane & 15, rb = (lane >> 4) << 3;
  float bv = bias[col0 + n];
  for (int r = 0; r < 8; ++r)
    out[(size_t)(row0 + rb + r) * N_SZ + col0 + n] = acc[r] + bv;
}

// ---------------- softmax + clip + minmax normalize (per row of 64) ----------------
__global__ void k_softmax_norm(const float* __restrict__ logits,
                               float* __restrict__ bvec) {
  __shared__ float red[64];
  int b = blockIdx.x, tid = threadIdx.x;
  float v = logits[(b << 6) + tid];
  red[tid] = v; __syncthreads();
  for (int s = 32; s > 0; s >>= 1) { if (tid < s) red[tid] = fmaxf(red[tid], red[tid + s]); __syncthreads(); }
  float mx = red[0]; __syncthreads();
  float e = __expf(v - mx);
  red[tid] = e; __syncthreads();
  for (int s = 32; s > 0; s >>= 1) { if (tid < s) red[tid] += red[tid + s]; __syncthreads(); }
  float p = e / red[0]; __syncthreads();
  p = fminf(fmaxf(p, 0.f), 1.f);
  red[tid] = p; __syncthreads();
  for (int s = 32; s > 0; s >>= 1) { if (tid < s) red[tid] = fminf(red[tid], red[tid + s]); __syncthreads(); }
  float mn = red[0]; __syncthreads();
  red[tid] = p; __syncthreads();
  for (int s = 32; s > 0; s >>= 1) { if (tid < s) red[tid] = fmaxf(red[tid], red[tid + s]); __syncthreads(); }
  float mxp = red[0];
  bvec[(b << 6) + tid] = (p - mn) / (mxp - mn + 1e-6f) + 1e-6f;
}

// ---------------- Q[b] = Qs[b]^T Qs[b] via WMMA on Qs^T rows ----------------
__global__ void k_qgram(const bf16_t* __restrict__ qst, float* __restrict__ Q) {
  int lane = threadIdx.x & 31;
  int wave = blockIdx.x * 8 + (threadIdx.x >> 5);
  int b = wave >> 4;
  int tile = wave & 15;
  int jt = tile >> 2, kt = tile & 3;
  const bf16_t* base = qst + ((size_t)b << 12);
  v8f acc;
  for (int r = 0; r < 8; ++r) acc[r] = 0.f;
  const bf16_t* A = base + (size_t)(jt << 4) * N_SZ;
  const bf16_t* Bw = base + (size_t)(kt << 4) * N_SZ;
  for (int kk = 0; kk < N_SZ; kk += 32)
    acc = wmma_bf16(ld_frag_a(A + kk, N_SZ, lane), ld_frag_b(Bw + kk, N_SZ, lane), acc);
  int n = lane & 15, rb = (lane >> 4) << 3;
  for (int r = 0; r < 8; ++r)
    Q[((size_t)b << 12) + (size_t)((jt << 4) + rb + r) * N_SZ + (kt << 4) + n] = acc[r];
}

// ---------------- damped Newton on KKT system, one block per batch ----------------
__global__ void k_newton(const float* __restrict__ Q, const float* __restrict__ bvec,
                         float* __restrict__ wout) {
  __shared__ float Km[65][66];
  __shared__ float yv[64], bb[64], byv[64], dv[65], red[64];
  __shared__ float sc[2];
  int b = blockIdx.x, tid = threadIdx.x;   // 64 threads
  const float* Qb = Q + ((size_t)b << 12);
  bb[tid] = bvec[(b << 6) + tid];
  red[tid] = bb[tid];
  __syncthreads();
  for (int s = 32; s > 0; s >>= 1) { if (tid < s) red[tid] += red[tid + s]; __syncthreads(); }
  float sumb = red[0];
  yv[tid] = __expf(0.1f / fmaxf(sumb, 1e-6f));
  if (tid == 0) sc[0] = 1.f;
  __syncthreads();

  for (int it = 0; it < 20; ++it) {
    float nu = sc[0];
    float yj = yv[tid];
    float byj = bb[tid] / yj;
    byv[tid] = byj;
    const float* Qr = Qb + tid * 64;
    float qy = 0.f;
    for (int k = 0; k < 64; ++k) qy += Qr[k] * yv[k];
    float r1 = 2.f * qy - nu * byj;
    red[tid] = bb[tid] * __logf(yj);
    for (int k = 0; k < 64; ++k) Km[tid][k] = 2.f * Qr[k];
    Km[tid][tid] += nu * byj / yj;
    Km[tid][64] = -byj;
    Km[tid][65] = -r1;
    __syncthreads();
    for (int s = 32; s > 0; s >>= 1) { if (tid < s) red[tid] += red[tid + s]; __syncthreads(); }
    float r2 = red[0] - 0.1f;
    Km[64][tid] = byv[tid];
    if (tid == 0) { Km[64][64] = 0.f; Km[64][65] = -r2; }
    __syncthreads();
    // Gaussian elimination, rows parallel over threads
    for (int col = 0; col < 65; ++col) {
      float piv = Km[col][col];
      piv = (fabsf(piv) < 1e-30f) ? 1e-30f : piv;
      for (int row = col + 1 + tid; row <= 64; row += 64) {
        float f = Km[row][col] / piv;
        for (int k = col; k <= 65; ++k) Km[row][k] -= f * Km[col][k];
      }
      __syncthreads();
    }
    if (tid == 0) {
      for (int row = 64; row >= 0; --row) {
        float s = Km[row][65];
        for (int k = row + 1; k <= 64; ++k) s -= Km[row][k] * dv[k];
        float pv = Km[row][row];
        pv = (fabsf(pv) < 1e-30f) ? 1e-30f : pv;
        dv[row] = s / pv;
      }
    }
    __syncthreads();
    float dy = dv[tid];
    float dnu = dv[64];
    red[tid] = (dy < 0.f) ? (-yj / dy) : __builtin_inff();
    __syncthreads();
    for (int s = 32; s > 0; s >>= 1) { if (tid < s) red[tid] = fminf(red[tid], red[tid + s]); __syncthreads(); }
    float alpha = fminf(1.f, 0.99f * red[0]);
    if (dnu < 0.f) alpha = fminf(alpha, 0.99f * (-nu / dnu));
    yv[tid] = yj + alpha * dy;
    if (tid == 0) sc[0] = nu + alpha * dnu;
    __syncthreads();
  }
  red[tid] = yv[tid];
  __syncthreads();
  for (int s = 32; s > 0; s >>= 1) { if (tid < s) red[tid] += red[tid + s]; __syncthreads(); }
  wout[(b << 6) + tid] = yv[tid] / red[0];
}

extern "C" void kernel_launch(void* const* d_in, const int* in_sizes, int n_in,
                              void* d_out, int out_size, void* d_ws, size_t ws_size,
                              hipStream_t stream) {
  const float* x    = (const float*)d_in[0];
  const float* qs   = (const float*)d_in[1];
  const float* w_ih = (const float*)d_in[2];
  const float* w_hh = (const float*)d_in[3];
  const float* b_ih = (const float*)d_in[4];
  const float* b_hh = (const float*)d_in[5];
  const float* fc1w = (const float*)d_in[6];
  const float* fc1b = (const float*)d_in[7];
  const float* fc2w = (const float*)d_in[8];
  const float* fc2b = (const float*)d_in[9];
  float* out = (float*)d_out;

  char* ws = (char*)d_ws;
  size_t off = 0;
  auto alloc = [&](size_t bytes) -> char* {
    char* p = ws + off;
    off += (bytes + 255) & ~(size_t)255;
    return p;
  };
  unsigned short* xbf    = (unsigned short*)alloc((size_t)B_SZ * T_SZ * D_SZ * 2);
  unsigned short* wihbf  = (unsigned short*)alloc((size_t)4 * H_SZ * D_SZ * 2);
  unsigned short* whhbf  = (unsigned short*)alloc((size_t)4 * H_SZ * H_SZ * 2);
  unsigned short* fc1wbf = (unsigned short*)alloc((size_t)F_SZ * H_SZ * 2);
  unsigned short* fc2wbf = (unsigned short*)alloc((size_t)N_SZ * F_SZ * 2);
  unsigned short* h0     = (unsigned short*)alloc((size_t)B_SZ * H_SZ * 2);
  unsigned short* h1     = (unsigned short*)alloc((size_t)B_SZ * H_SZ * 2);
  float*          cbuf   = (float*)alloc((size_t)B_SZ * H_SZ * 4);
  unsigned short* b1bf   = (unsigned short*)alloc((size_t)B_SZ * F_SZ * 2);
  float*          logit  = (float*)alloc((size_t)B_SZ * N_SZ * 4);
  float*          bvec   = (float*)alloc((size_t)B_SZ * N_SZ * 4);
  unsigned short* qstbf  = (unsigned short*)alloc((size_t)B_SZ * N_SZ * N_SZ * 2);
  float*          Qf     = (float*)alloc((size_t)B_SZ * N_SZ * N_SZ * 4);

  const int nx   = B_SZ * T_SZ * D_SZ;      // 4194304
  const int nwih = 4 * H_SZ * D_SZ;         // 131072
  const int nwhh = 4 * H_SZ * H_SZ;         // 1048576
  const int nf1  = F_SZ * H_SZ;             // 524288
  const int nf2  = N_SZ * F_SZ;             // 65536
  const int nqs  = B_SZ * N_SZ * N_SZ;      // 1048576
  const int nst  = B_SZ * H_SZ;             // 131072

  k_f32_to_bf16<<<(nx + 255) / 256, 256, 0, stream>>>(x, xbf, nx);
  k_f32_to_bf16<<<(nwih + 255) / 256, 256, 0, stream>>>(w_ih, wihbf, nwih);
  k_f32_to_bf16<<<(nwhh + 255) / 256, 256, 0, stream>>>(w_hh, whhbf, nwhh);
  k_f32_to_bf16<<<(nf1 + 255) / 256, 256, 0, stream>>>(fc1w, fc1wbf, nf1);
  k_f32_to_bf16<<<(nf2 + 255) / 256, 256, 0, stream>>>(fc2w, fc2wbf, nf2);
  k_qs_transpose_bf16<<<(nqs + 255) / 256, 256, 0, stream>>>(qs, qstbf, nqs);
  k_init_state<<<(nst + 255) / 256, 256, 0, stream>>>(h0, cbuf, nst);

  for (int t = 0; t < T_SZ; ++t) {
    const unsigned short* hin = (t & 1) ? h1 : h0;
    unsigned short* hout = (t & 1) ? h0 : h1;
    k_lstm_step<<<64, 256, 0, stream>>>((const bf16_t*)xbf, (const bf16_t*)wihbf,
                                        (const bf16_t*)whhbf, b_ih, b_hh,
                                        (const bf16_t*)hin, (bf16_t*)hout, cbuf, t);
  }
  // T=256 (even): last step t=255 wrote into h0
  k_fc1<<<128, 256, 0, stream>>>((const bf16_t*)h0, (const bf16_t*)fc1wbf, fc1b, b1bf);
  k_fc2<<<8, 256, 0, stream>>>((const bf16_t*)b1bf, (const bf16_t*)fc2wbf, fc2b, logit);
  k_softmax_norm<<<256, 64, 0, stream>>>(logit, bvec);
  k_qgram<<<512, 256, 0, stream>>>((const bf16_t*)qstbf, Qf);
  k_newton<<<256, 64, 0, stream>>>(Qf, bvec, out);
}
